// PhyloNeighbours_8461085573180
// MI455X (gfx1250) — matrix operand order repaired and checked by reading, not compiled, not run on hardware
//
#include <hip/hip_runtime.h>

// ---------------------------------------------------------------------------
// PhyloNeighbours on MI455X (gfx1250, wave32)
//
//  inputs      : (B=64, C=4, F=8192, 1) f32
//  coordinates : (d=64, 1, F=8192) f32
//  out         : (64, 4, F*8, 1) f32  = 16.78M elements (67 MB write floor)
//
//  Fused pipeline:
//   1) prep   : coordsT[f][k] = bf16(coords[k][f]);  sq[f] = sum_k coords[k][f]^2
//   2) topk   : per 16-row block, stream 32-col tiles; Gram via
//               v_wmma_f32_16x16x32_bf16 (4 WMMAs / iter, K=64, A reused);
//               dist2 = sq[m]+sq[n]-2G; running top-8 per row.
//               LDS staging is wave-private -> wave-local s_wait_dscnt fence,
//               no workgroup barriers (per-wave DS ops are in-order).
//   3) gather : out[bc*65536+g] = in[bc*8192 + (g==0 ? 0 : neigh[g])]
// ---------------------------------------------------------------------------

typedef __attribute__((ext_vector_type(16))) __bf16 v16bf;
typedef __attribute__((ext_vector_type(8)))  float  v8f;

#define F_DIM 8192
#define K_DIM 64
#define NB    8

// workspace layout (bytes)
#define WS_SQ_OFF    0                           // float[8192]            32 KB
#define WS_CT_OFF    32768                       // bf16 [8192*64]          1 MB
#define WS_NEIGH_OFF (32768 + F_DIM * K_DIM * 2) // int[8192*8]           256 KB

// Wave-local LDS fence: per-wave DS ops are processed in order (ISA 11.x);
// this just drains DScnt and stops the compiler reordering LDS ops across it.
#define WAVE_LDS_FENCE() asm volatile("s_wait_dscnt 0" ::: "memory")

__device__ __forceinline__ unsigned short f32_to_bf16(float x) {
    unsigned int u = __float_as_uint(x);
    u += 0x7fffu + ((u >> 16) & 1u);             // round-to-nearest-even
    return (unsigned short)(u >> 16);
}

// ---------------------------------------------------------------------------
// Kernel 1: transpose+convert coordinates, compute squared norms
// ---------------------------------------------------------------------------
__global__ void phylo_prep_kernel(const float* __restrict__ coords,
                                  float* __restrict__ sq,
                                  unsigned short* __restrict__ ct) {
    int f = blockIdx.x * 256 + threadIdx.x;      // 8192 threads
    float acc = 0.0f;
    #pragma unroll
    for (int k = 0; k < K_DIM; ++k) {
        float v = coords[k * F_DIM + f];         // coalesced across lanes per k
        acc += v * v;
        ct[f * K_DIM + k] = f32_to_bf16(v);
    }
    sq[f] = acc;
}

// ---------------------------------------------------------------------------
// Kernel 2: fused Gram (WMMA bf16) + per-row running top-8
//   block = 128 threads = 4 waves; wave w owns row block (blockIdx.x*4+w)*16
//   Each iteration handles a 16x32 distance tile (two 16x16 WMMA tiles).
// ---------------------------------------------------------------------------
__launch_bounds__(128)
__global__ void phylo_topk_kernel(const unsigned int* __restrict__ ctd, // coordsT dwords (32/row)
                                  const float* __restrict__ sq,
                                  int* __restrict__ neigh) {
    __shared__ float sdist[4][16 * 33];          // padded 16x32 dist tile per wave
    __shared__ float smrg[4][16][32];            // merge staging: 2 x (8 dist + 8 idx) per row

    const int lane = threadIdx.x & 31;
    const int wv   = threadIdx.x >> 5;
    const int half = lane >> 4;                  // 0: lanes 0-15, 1: lanes 16-31
    const int l16  = lane & 15;
    const int m0   = (blockIdx.x * 4 + wv) * 16;

    union U8 { unsigned int u[8]; v16bf v; };

    // ---- A operands (rows m0..m0+15, K=0..63), fixed across column loop ----
    // A 16x32 bf16 layout: lanes0-15 -> dwords {db..db+3, db+8..db+11}, db = half*4
    U8 a0, a1;
    {
        const unsigned int* rowp = ctd + (m0 + l16) * 32;
        const int db = half * 4;
        #pragma unroll
        for (int i = 0; i < 4; ++i) { a0.u[i] = rowp[db + i];      a0.u[4 + i] = rowp[db + 8 + i]; }
        #pragma unroll
        for (int i = 0; i < 4; ++i) { a1.u[i] = rowp[db + 16 + i]; a1.u[4 + i] = rowp[db + 24 + i]; }
    }

    // sq for my accumulator rows: c[r] holds M = r + 8*half
    float sqm[8];
    #pragma unroll
    for (int r = 0; r < 8; ++r) sqm[r] = sq[m0 + r + 8 * half];

    // running top-8 (ascending): lane handles row l16, cols half*8..half*8+7 of each tile
    float bd[8]; int bi[8];
    #pragma unroll
    for (int k = 0; k < 8; ++k) { bd[k] = 3.4e38f; bi[k] = 0; }

    float* sld = sdist[wv];
    const int cb = half * 8;

    for (int n0 = 0; n0 < F_DIM; n0 += 32) {
        // ---- B operands for two column tiles (n0, n0+16) ----
        // B 32x16 bf16 layout (matches A): dwords {db..db+3, db+8..db+11}, db = half*4
        U8 b0[2], b1[2];
        #pragma unroll
        for (int t = 0; t < 2; ++t) {
            const unsigned int* rowp = ctd + (n0 + 16 * t + l16) * 32;
            const int db = half * 4;
            #pragma unroll
            for (int i = 0; i < 8; ++i) b0[t].u[i] = rowp[db + (i & 3) + ((i >> 2) << 3)];
            #pragma unroll
            for (int i = 0; i < 8; ++i) b1[t].u[i] = rowp[16 + db + (i & 3) + ((i >> 2) << 3)];
        }

        v8f c0 = {}, c1 = {};
        c0 = __builtin_amdgcn_wmma_f32_16x16x32_bf16(false, a0.v, false, b0[0].v, (short)0, c0, false, false);
        c0 = __builtin_amdgcn_wmma_f32_16x16x32_bf16(false, a1.v, false, b1[0].v, (short)0, c0, false, false);
        c1 = __builtin_amdgcn_wmma_f32_16x16x32_bf16(false, a0.v, false, b0[1].v, (short)0, c1, false, false);
        c1 = __builtin_amdgcn_wmma_f32_16x16x32_bf16(false, a1.v, false, b1[1].v, (short)0, c1, false, false);

        float sqn0 = sq[n0 + l16];               // column norms, N = l16 of each tile
        float sqn1 = sq[n0 + 16 + l16];

        WAVE_LDS_FENCE();                        // guard vs. previous iteration's reads
        #pragma unroll
        for (int r = 0; r < 8; ++r) {
            float d20 = fmaxf(sqm[r] + sqn0 - 2.0f * c0[r], 0.0f);  // matches max(d,0)
            float d21 = fmaxf(sqm[r] + sqn1 - 2.0f * c1[r], 0.0f);
            sld[(r + 8 * half) * 33 + l16]      = d20;
            sld[(r + 8 * half) * 33 + 16 + l16] = d21;
        }
        WAVE_LDS_FENCE();                        // stores visible to same-wave readers

        // ---- scan 16 candidates for row l16 (ascending column order) ----
        #pragma unroll
        for (int t = 0; t < 2; ++t) {
            #pragma unroll
            for (int j = 0; j < 8; ++j) {
                float dv = sld[l16 * 33 + 16 * t + cb + j];
                int   ci = n0 + 16 * t + cb + j;
                if (dv < bd[7]) {                // strict < : lowest index wins ties
                    float cd = dv; int cx = ci;
                    #pragma unroll
                    for (int k = 0; k < 8; ++k) {
                        bool sw = cd < bd[k];
                        float td = sw ? bd[k] : cd;   int ti = sw ? bi[k] : cx;
                        bd[k]    = sw ? cd    : bd[k]; bi[k] = sw ? cx : bi[k];
                        cd = td; cx = ti;
                    }
                }
            }
        }
    }

    // ---- merge the two per-row half-lists (tie-break on smaller index) ----
    WAVE_LDS_FENCE();                            // last scan reads done before reuse
    #pragma unroll
    for (int k = 0; k < 8; ++k) {
        smrg[wv][l16][half * 16 + k]     = bd[k];
        smrg[wv][l16][half * 16 + 8 + k] = __int_as_float(bi[k]);
    }
    WAVE_LDS_FENCE();                            // wave-private staging visible
    if (lane < 16) {
        const float* mA = &smrg[wv][l16][0];
        const float* mB = &smrg[wv][l16][16];
        int ia = 0, ib = 0;
        int* orow = neigh + (m0 + l16) * NB;
        #pragma unroll
        for (int t = 0; t < 8; ++t) {
            float da  = mA[ia],                  dbv = mB[ib];
            int   ja  = __float_as_int(mA[8 + ia]);
            int   jb  = __float_as_int(mB[8 + ib]);
            bool  tb  = (dbv < da) || (dbv == da && jb < ja);
            orow[t] = tb ? jb : ja;
            ia += tb ? 0 : 1;
            ib += tb ? 1 : 0;
        }
    }
}

// ---------------------------------------------------------------------------
// Kernel 3: fused flat_idx build + gather   (flat_idx[0] hardcoded to 0)
// ---------------------------------------------------------------------------
__global__ void phylo_gather_kernel(const float* __restrict__ in,
                                    const int* __restrict__ neigh,
                                    float* __restrict__ out) {
    int g   = blockIdx.x * 256 + threadIdx.x;    // 16,777,216 threads
    int off = g & 65535;                         // position along F*K axis
    int bc  = g >> 16;                           // which (b,c) slice
    int src = (off == 0) ? 0 : neigh[off];       // coalesced idx load, cached input slice
    out[g] = in[bc * F_DIM + src];               // coalesced store
}

// ---------------------------------------------------------------------------
extern "C" void kernel_launch(void* const* d_in, const int* in_sizes, int n_in,
                              void* d_out, int out_size, void* d_ws, size_t ws_size,
                              hipStream_t stream) {
    (void)in_sizes; (void)n_in; (void)out_size; (void)ws_size;

    const float* inputs = (const float*)d_in[0];   // (64,4,8192,1)
    const float* coords = (const float*)d_in[1];   // (64,1,8192)
    float* out = (float*)d_out;

    char* ws = (char*)d_ws;
    float*          sq    = (float*)(ws + WS_SQ_OFF);
    unsigned short* ct    = (unsigned short*)(ws + WS_CT_OFF);
    unsigned int*   ctd   = (unsigned int*)(ws + WS_CT_OFF);
    int*            neigh = (int*)(ws + WS_NEIGH_OFF);

    // 1) transpose/convert + norms: 8192 threads
    phylo_prep_kernel<<<F_DIM / 256, 256, 0, stream>>>(coords, sq, ct);

    // 2) fused WMMA distance + top-8: 512 row-blocks, 4 waves (one block each) per CTA
    phylo_topk_kernel<<<(F_DIM / 16) / 4, 128, 0, stream>>>(ctd, sq, neigh);

    // 3) gather into output: 64*4*65536 elements
    phylo_gather_kernel<<<(64 * 4 * F_DIM * NB) / 256, 256, 0, stream>>>(inputs, neigh, out);
}
